// ResGCN_23192823398695
// MI455X (gfx1250) — matrix operand order repaired and checked by reading, not compiled, not run on hardware
//
#include <hip/hip_runtime.h>
#include <cstdint>
#include <cstddef>

typedef _Float16 v16h __attribute__((ext_vector_type(16)));
typedef float    v8f  __attribute__((ext_vector_type(8)));

#define DFEAT 128

// ---------------- utility kernels ----------------

__global__ void k_fill_f32(float* __restrict__ p, float v, int n) {
  int i = blockIdx.x * blockDim.x + threadIdx.x;
  if (i < n) p[i] = v;
}

__global__ void k_zero_f32v4(float4* __restrict__ p, int n4) {
  int i = blockIdx.x * blockDim.x + threadIdx.x;
  if (i < n4) p[i] = make_float4(0.f, 0.f, 0.f, 0.f);
}

// deg[dst[e]] += 1  (deg pre-initialized to 1.0 for the self-loop)
__global__ void k_degree(const int* __restrict__ dst, float* __restrict__ deg, int E) {
  int e = blockIdx.x * blockDim.x + threadIdx.x;
  if (e < E) atomicAdd(&deg[dst[e]], 1.0f);
}

// deg -> dinv in place; deg >= 1 always (self loops), so plain rsqrt
__global__ void k_rsqrt_inplace(float* __restrict__ p, int n) {
  int i = blockIdx.x * blockDim.x + threadIdx.x;
  if (i < n) p[i] = rsqrtf(p[i]);
}

// ---------------- W pre-swizzle into WMMA B-fragment layout ----------------
// Wh[((t*4 + kb)*32 + lane)*16 + h] = (f16) W[(32*kb + (lane>=16?16:0) + h)*128 + 16*t + (lane&15)]
// so each wave's B fragment for (col-tile t, k-step kb) is 32 contiguous bytes
// per lane -> two global_load_b128, no cvt in the GEMM inner loop.
__global__ void k_swizzleW(const float* __restrict__ W, _Float16* __restrict__ Wh) {
  int i = blockIdx.x * blockDim.x + threadIdx.x;      // 8*4*32*16 = 16384
  if (i >= 8 * 4 * 32 * 16) return;
  const int h    = i & 15;
  const int lane = (i >> 4) & 31;
  const int kb   = (i >> 9) & 3;
  const int t    = i >> 11;
  const int m    = lane & 15;
  const int kk   = 32 * kb + ((lane >> 4) ? 16 : 0) + h;
  Wh[i] = (_Float16)W[(size_t)kk * DFEAT + 16 * t + m];
}

// ---------------- WMMA GEMM: T[N,128] = X[N,128] @ W[128,128] ----------------
// Block = 256 threads = 8 waves; blockIdx.x = 16-row tile; wave w owns output
// columns [16w, 16w+16). The 16x128 A tile is staged once per block into LDS
// in the per-lane A-fragment layout (f16, k = 16*(l/16) + j), then every wave reads it with
// ds_load_b128. B comes pre-swizzled (k_swizzleW). 4 k-steps of
// v_wmma_f32_16x16x32_f16, f32 accumulation.
__global__ void __launch_bounds__(256)
k_gemm_xw(const float* __restrict__ X, const _Float16* __restrict__ Wh,
          float* __restrict__ T, int nrows) {
  __shared__ __align__(32) _Float16 Ash[4 * 32 * 16];   // 4 k-steps, per-lane v16h

  const int tid  = threadIdx.x;
  const int wave = tid >> 5;
  const int lane = tid & 31;
  const int m    = lane & 15;
  const bool hi  = lane >= 16;
  const int row0 = blockIdx.x * 16;
  const int col0 = wave * 16;

  // ---- cooperative fill of A tile in fragment layout (2048 f16, 8 per thread)
  // element (kb, l, j): K = 32*kb + (l>=16?16:0) + j
#pragma unroll
  for (int u = 0; u < 8; ++u) {
    const int idx  = tid + u * 256;
    const int kb   = idx >> 9;
    const int rem  = idx & 511;
    const int l    = rem >> 4;
    const int j    = rem & 15;
    const int base = (l >> 4) ? 16 : 0;
    const int koff = base + j;
    int r = row0 + (l & 15);
    if (r >= nrows) r = nrows - 1;                       // clamp (N % 16 == 0 in practice)
    Ash[idx] = (_Float16)X[(size_t)r * DFEAT + (kb << 5) + koff];
  }
  __syncthreads();

  v8f acc = {};
#pragma unroll
  for (int kb = 0; kb < 4; ++kb) {
    const v16h a = *(const v16h*)&Ash[(kb * 32 + lane) * 16];
    const v16h b = *(const v16h*)&Wh[(size_t)(((wave * 4) + kb) * 32 + lane) * 16];
    // (neg_a, A, neg_b, B, c_mod, C, reuse_a, reuse_b)
    acc = __builtin_amdgcn_wmma_f32_16x16x32_f16(false, a, false, b,
                                                 (short)0, acc, false, false);
  }

  // ---- C/D store: VGPR r -> M = r (+8 for hi lanes), N = lane&15 ----
  // Scalar (block-uniform) bounds test: full tiles take the unguarded path.
  const int outcol  = col0 + m;
  const int rowbase = row0 + (hi ? 8 : 0);
  float* __restrict__ tp = T + (size_t)rowbase * DFEAT + outcol;
  if (row0 + 16 <= nrows) {
#pragma unroll
    for (int r = 0; r < 8; ++r) tp[(size_t)r * DFEAT] = acc[r];
  } else {
#pragma unroll
    for (int r = 0; r < 8; ++r) {
      if (rowbase + r < nrows) tp[(size_t)r * DFEAT] = acc[r];
    }
  }
}

// ---------------- edge scatter: G[dst] += dinv[src]*dinv[dst] * T[src] --------
// One wave per edge; lane l handles features [4l, 4l+4) via float4 load and
// four global f32 atomics. Edge scalars are pulled into SGPRs with
// readfirstlane so addressing runs on the SALU. The 51MB feature matrix is
// resident in the 192MB L2, so gather + atomics hit cache.
__global__ void __launch_bounds__(256)
k_scatter(const int* __restrict__ src, const int* __restrict__ dst,
          const float* __restrict__ dinv, const float* __restrict__ T,
          float* __restrict__ G, int E) {
  const int wv = __builtin_amdgcn_readfirstlane(threadIdx.x >> 5);
  const int e  = blockIdx.x * 8 + wv;
  if (e >= E) return;
  const int lane = threadIdx.x & 31;

  const int s = __builtin_amdgcn_readfirstlane(src[e]);
  const int d = __builtin_amdgcn_readfirstlane(dst[e]);
  const float nrm = dinv[s] * dinv[d];

  const float4 v = ((const float4*)(T + (size_t)s * DFEAT))[lane];
  float* gp = G + (size_t)d * DFEAT + lane * 4;
  atomicAdd(gp + 0, v.x * nrm);
  atomicAdd(gp + 1, v.y * nrm);
  atomicAdd(gp + 2, v.z * nrm);
  atomicAdd(gp + 3, v.w * nrm);
}

// ---------------- epilogue: out = act(G + dinv^2*T + b [+ res]) --------------
// Fuses the self-loop edge (norm = dinv[n]^2), bias, residual and ReLU.
// Elementwise, so out may alias res (in-place layer-1 update is safe).
__global__ void __launch_bounds__(256)
k_epilogue(const float* __restrict__ G, const float* __restrict__ T,
           const float* __restrict__ dinv, const float* __restrict__ bias,
           const float* __restrict__ res, float* __restrict__ out,
           int n_elems, int do_relu) {
  int i = blockIdx.x * blockDim.x + threadIdx.x;
  if (i >= n_elems) return;
  const int node = i >> 7;      // DFEAT == 128
  const int f    = i & (DFEAT - 1);
  const float di = dinv[node];
  float v = G[i] + di * di * T[i] + bias[f];
  if (res != nullptr) v += res[i];
  if (do_relu) v = fmaxf(v, 0.f);
  out[i] = v;
}

// ---------------- driver ----------------

extern "C" void kernel_launch(void* const* d_in, const int* in_sizes, int n_in,
                              void* d_out, int out_size, void* d_ws, size_t ws_size,
                              hipStream_t stream) {
  const float* x  = (const float*)d_in[0];
  const int*   ei = (const int*)d_in[1];   // [2, E] int (src row, then dst row)
  const float* W0 = (const float*)d_in[2];
  const float* b0 = (const float*)d_in[3];
  const float* W1 = (const float*)d_in[4];
  const float* b1 = (const float*)d_in[5];
  const float* W2 = (const float*)d_in[6];
  const float* b2 = (const float*)d_in[7];
  float* out = (float*)d_out;

  const int N = in_sizes[0] / DFEAT;   // 100000
  const int E = in_sizes[1] / 2;       // 3200000
  const int NF = N * DFEAT;            // 12.8M elems

  const int* src = ei;
  const int* dst = ei + E;

  // carve workspace (256B aligned chunks)
  size_t off = 0;
  char* base = (char*)d_ws;
  auto carve = [&](size_t bytes) -> void* {
    void* p = base + off;
    off += (bytes + 255) & ~(size_t)255;
    return p;
  };
  float*    dinv = (float*)carve((size_t)N * sizeof(float));     // degree -> dinv
  float*    H    = (float*)carve((size_t)NF * sizeof(float));    // current node features
  float*    T    = (float*)carve((size_t)NF * sizeof(float));    // H @ W (pre-aggregation)
  float*    G    = (float*)carve((size_t)NF * sizeof(float));    // aggregated messages
  _Float16* Wh   = (_Float16*)carve((size_t)8 * 4 * 32 * 16 * sizeof(_Float16)); // swizzled W

  const int TPB = 256;
  const int gN   = (N + TPB - 1) / TPB;
  const int gE   = (E + TPB - 1) / TPB;
  const int gNF  = (NF + TPB - 1) / TPB;
  const int gNF4 = ((NF / 4) + TPB - 1) / TPB;
  const int gGemm = (N + 15) / 16;                       // 16-row tiles
  const int gScat = (E + 7) / 8;                         // 8 waves/block, wave/edge
  const int gSwz  = (8 * 4 * 32 * 16 + TPB - 1) / TPB;   // 64 blocks

  // ---- gcn_norm: deg = 1 (self-loop) + scatter-count over dst; dinv = rsqrt ----
  k_fill_f32<<<gN, TPB, 0, stream>>>(dinv, 1.0f, N);
  k_degree<<<gE, TPB, 0, stream>>>(dst, dinv, E);
  k_rsqrt_inplace<<<gN, TPB, 0, stream>>>(dinv, N);

  // ---- layer 0: H = relu(Agg(x@W0) + dinv^2*(x@W0) + b0) ----
  k_swizzleW<<<gSwz, TPB, 0, stream>>>(W0, Wh);
  k_gemm_xw<<<gGemm, TPB, 0, stream>>>(x, Wh, T, N);
  k_zero_f32v4<<<gNF4, TPB, 0, stream>>>((float4*)G, NF / 4);
  k_scatter<<<gScat, TPB, 0, stream>>>(src, dst, dinv, T, G, E);
  k_epilogue<<<gNF, TPB, 0, stream>>>(G, T, dinv, b0, nullptr, H, NF, 1);

  // ---- layer 1 (residual): H = relu(Agg(H@W1) + dinv^2*(H@W1) + b1 + H) ----
  k_swizzleW<<<gSwz, TPB, 0, stream>>>(W1, Wh);
  k_gemm_xw<<<gGemm, TPB, 0, stream>>>(H, Wh, T, N);
  k_zero_f32v4<<<gNF4, TPB, 0, stream>>>((float4*)G, NF / 4);
  k_scatter<<<gScat, TPB, 0, stream>>>(src, dst, dinv, T, G, E);
  k_epilogue<<<gNF, TPB, 0, stream>>>(G, T, dinv, b1, H, H, NF, 1);  // in-place ok

  // ---- layer 2: out = Agg(H@W2) + dinv^2*(H@W2) + b2 ----
  k_swizzleW<<<gSwz, TPB, 0, stream>>>(W2, Wh);
  k_gemm_xw<<<gGemm, TPB, 0, stream>>>(H, Wh, T, N);
  k_zero_f32v4<<<gNF4, TPB, 0, stream>>>((float4*)G, NF / 4);
  k_scatter<<<gScat, TPB, 0, stream>>>(src, dst, dinv, T, G, E);
  k_epilogue<<<gNF, TPB, 0, stream>>>(G, T, dinv, b2, nullptr, out, NF, 0);
}
